// TextVQVAE_48911087566896
// MI455X (gfx1250) — compile-verified, matrix operand order checked by reading, not run on hardware
//
#include <hip/hip_runtime.h>
#include <hip/hip_bf16.h>
#include <stdint.h>

// ---------------------------------------------------------------------------
// CDNA5 (gfx1250) wave32 WMMA implementation of the TextVQVAE forward pass.
// All large matmuls use v_wmma_f32_16x16x32_bf16 (fp32 accumulate).
// GEMM B-tiles are staged to LDS with global_load_async_to_lds_b128 (ASYNCcnt).
// ---------------------------------------------------------------------------

typedef __bf16 bf16_t;
typedef __attribute__((ext_vector_type(16))) __bf16 v16bf;
typedef __attribute__((ext_vector_type(8)))  __bf16 bf16x8;
typedef __attribute__((ext_vector_type(8)))  float  v8f;

#define DEV __device__ __forceinline__

#if __has_builtin(__builtin_amdgcn_s_wait_asynccnt)
#define WAIT_ASYNC(n) __builtin_amdgcn_s_wait_asynccnt(n)
#else
#define WAIT_ASYNC(n) asm volatile("s_wait_asynccnt %0" :: "i"(n) : "memory")
#endif

DEV int lane_id() { return threadIdx.x & 31; }

// One lane moves 16 bytes global -> LDS through the async copy unit.
DEV void async_copy_b128(unsigned lds_off_bytes, const void* gptr) {
  asm volatile("global_load_async_to_lds_b128 %0, %1, off"
               :: "v"(lds_off_bytes), "v"((unsigned long long)(uintptr_t)gptr)
               : "memory");
}

DEV v8f wmma_bf16(v16bf a, v16bf b, v8f c) {
  // D = A(16x32) * B(32x16) + C(16x16), fp32 accumulate
  return __builtin_amdgcn_wmma_f32_16x16x32_bf16(
      /*neg_a=*/false, a, /*neg_b=*/false, b,
      /*c_mod=*/(short)0, c, /*reuse_a=*/false, /*reuse_b=*/false);
}

// 16-bit A/B operand fragment: lane l holds row (l&15); K-halves split by (l>>4).
// Lane reads 8 contiguous bf16 at +0 and 8 more at +16 elements (two b128 loads).
DEV v16bf frag_load(const bf16_t* base, int stride, int row) {
  int l = lane_id();
  const bf16_t* p = base + (size_t)(row + (l & 15)) * stride + ((l >> 4) << 3);
  union { v16bf v; bf16x8 h[2]; } u;
  u.h[0] = *reinterpret_cast<const bf16x8*>(p);
  u.h[1] = *reinterpret_cast<const bf16x8*>(p + 16);
  return u.v;
}

DEV v16bf frag_load_cmin(const bf16_t* base, int stride, int row, int minRow) {
  int l = lane_id();
  int r = row + (l & 15); if (r < minRow) r = minRow;
  const bf16_t* p = base + (size_t)r * stride + ((l >> 4) << 3);
  union { v16bf v; bf16x8 h[2]; } u;
  u.h[0] = *reinterpret_cast<const bf16x8*>(p);
  u.h[1] = *reinterpret_cast<const bf16x8*>(p + 16);
  return u.v;
}

DEV float wave_sum(float v) {
  for (int off = 16; off; off >>= 1) v += __shfl_xor(v, off, 32);
  return v;
}
DEV float wave_max(float v) {
  for (int off = 16; off; off >>= 1) v = fmaxf(v, __shfl_xor(v, off, 32));
  return v;
}

// ---------------------------------------------------------------------------
// Tiled GEMM: C[M,N] (fp32) = A[M,K](bf16,row-major) * Bt[N,K](bf16)^T
// Block = 256 threads = 8 waves in 2x4; wave tile 32x32 (2x2 WMMA tiles);
// block tile 64x128. M must be a multiple of 64; N edge handled by clamping.
// B tile (128x32 bf16, 8KB) is double-buffered in LDS via async copies:
// each wave issues 2x global_load_async_to_lds_b128 per K-step (16B/lane),
// retires the previous tile with s_wait_asynccnt 2 while the next is in
// flight, and reads WMMA B-operands from LDS. A-operands load directly from
// global (b128) with global_prefetch_b8 lookahead.
// ---------------------------------------------------------------------------
__global__ void __launch_bounds__(256)
k_gemm(const bf16_t* __restrict__ A, const bf16_t* __restrict__ Bt,
       float* __restrict__ C, const float* __restrict__ bias,
       int N, int K, int lda, int ldb, int ldc, int addRes)
{
  __shared__ __align__(16) bf16_t sB[2][128 * 32];

  int tid  = threadIdx.x;
  int w    = tid >> 5;
  int wm   = w >> 2, wn = w & 3;
  int m0   = blockIdx.y * 64 + wm * 32;
  int nblk = blockIdx.x * 128;

  int l = tid & 31;
  // async-staging coordinates: wave w stages rows [w*16, w*16+16) of the tile;
  // one instruction covers 8 rows (4 lanes x 16B per 64B row).
  int srow = w * 16 + (l >> 2);
  int scol = (l & 3) * 8;

  auto stage = [&](int k, int buf) {
    for (int q = 0; q < 2; q++) {
      int row = srow + q * 8;
      int gr  = nblk + row; if (gr > N - 1) gr = N - 1;
      const bf16_t* g = Bt + (size_t)gr * ldb + k + scol;
      unsigned loff = (unsigned)(uintptr_t)&sB[buf][row * 32 + scol];
      async_copy_b128(loff, g);
    }
  };

  stage(0, 0);

  v8f acc00 = {}, acc01 = {}, acc10 = {}, acc11 = {};
  int it = 0;
  for (int k = 0; k < K; k += 32, it++) {
    int cur = it & 1;
    bool more = (k + 32 < K);
    if (more) {
      stage(k + 32, cur ^ 1);
      __builtin_prefetch(A + (size_t)m0 * lda + k + 32, 0, 1);
      WAIT_ASYNC(2);   // previous tile's 2 ops retired; next tile in flight
    } else {
      WAIT_ASYNC(0);
    }
    __syncthreads();   // all waves' async writes for tile `cur` visible

    v16bf a0 = frag_load(A + k, lda, m0);
    v16bf a1 = frag_load(A + k, lda, m0 + 16);
    const bf16_t* bp = &sB[cur][0];
    v16bf b0 = frag_load(bp, 32, wn * 32);
    v16bf b1 = frag_load(bp, 32, wn * 32 + 16);
    acc00 = wmma_bf16(a0, b0, acc00);
    acc01 = wmma_bf16(a0, b1, acc01);
    acc10 = wmma_bf16(a1, b0, acc10);
    acc11 = wmma_bf16(a1, b1, acc11);

    __syncthreads();   // tile `cur` fully consumed before it is overwritten
  }

  int n0 = nblk + wn * 32;
  int nl = l & 15, mo = (l >> 4) << 3;
  v8f accs[2][2] = {{acc00, acc01}, {acc10, acc11}};
  for (int i = 0; i < 2; i++)
    for (int j = 0; j < 2; j++) {
      int n = n0 + j * 16 + nl;
      if (n >= N) continue;
      float bv = bias ? bias[n] : 0.0f;
      for (int vv = 0; vv < 8; vv++) {
        int m = m0 + i * 16 + mo + vv;
        size_t idx = (size_t)m * ldc + n;
        float val = accs[i][j][vv] + bv;
        if (addRes) val += C[idx];
        C[idx] = val;
      }
    }
}

// ---------------------------------------------------------------------------
// Windowed causal attention, one block per (window, head, batch).
// qw[32,64] x kk[64,64]^T -> softmax -> x vv[64,64].  All matmuls via WMMA.
// ---------------------------------------------------------------------------
__global__ void __launch_bounds__(256)
k_attention(const bf16_t* __restrict__ QR, const bf16_t* __restrict__ KR,
            const bf16_t* __restrict__ VR, bf16_t* __restrict__ OH)
{
  __shared__ __align__(16) float  sSim[32 * 64];
  __shared__ __align__(16) bf16_t sP[32 * 64];
  __shared__ __align__(16) bf16_t sVT[64 * 64];

  int w = blockIdx.x, h = blockIdx.y, b = blockIdx.z;
  int wid = threadIdx.x >> 5;
  int wm = wid >> 2, wn = wid & 3;
  int bbase = b * 4096;
  int qbase = bbase + w * 32;
  int kbase = bbase + (w - 1) * 32;  // may dip below bbase for w==0 (masked)

  // stage V^T into LDS: sVT[d][key]
  for (int i = 0; i < 16; i++) {
    int e = threadIdx.x * 16 + i;
    int key = e >> 6, d = e & 63;
    int t = kbase + key; if (t < bbase) t = bbase;
    sVT[d * 64 + key] = VR[(size_t)t * 512 + h * 64 + d];
  }

  // sim = q @ k^T  (wave owns one 16x16 tile of the 32x64 result)
  v8f acc = {};
  for (int k = 0; k < 64; k += 32) {
    v16bf a  = frag_load(QR + h * 64 + k, 512, qbase + wm * 16);
    v16bf bf = frag_load_cmin(KR + h * 64 + k, 512, kbase + wn * 16, bbase);
    acc = wmma_bf16(a, bf, acc);
  }
  {
    int l = lane_id();
    int j = wn * 16 + (l & 15);
    int mo = (l >> 4) << 3;
    for (int vv = 0; vv < 8; vv++) {
      int i = wm * 16 + mo + vv;
      bool m = (j <= i + 32) && (w > 0 || j >= 32);
      sSim[i * 64 + j] = m ? acc[vv] * 8.0f : -1e9f;
    }
  }
  __syncthreads();

  // softmax per query row (32 rows handled by first wave)
  if (threadIdx.x < 32) {
    int i = threadIdx.x;
    float mx = -3.4e38f;
    for (int j = 0; j < 64; j++) mx = fmaxf(mx, sSim[i * 64 + j]);
    float ev[64]; float s = 0.0f;
    for (int j = 0; j < 64; j++) { ev[j] = __expf(sSim[i * 64 + j] - mx); s += ev[j]; }
    float inv = 1.0f / s;
    for (int j = 0; j < 64; j++) sP[i * 64 + j] = (bf16_t)(ev[j] * inv);
  }
  __syncthreads();

  // out = P @ V  (B operand is V^T staged in LDS; A operand P from LDS)
  v8f o = {};
  for (int k = 0; k < 64; k += 32) {
    v16bf a  = frag_load((const bf16_t*)sP + k, 64, wm * 16);
    v16bf bf = frag_load((const bf16_t*)sVT + k, 64, wn * 16);
    o = wmma_bf16(a, bf, o);
  }
  {
    int l = lane_id();
    int d = wn * 16 + (l & 15);
    int mo = (l >> 4) << 3;
    for (int vv = 0; vv < 8; vv++) {
      int i = wm * 16 + mo + vv;
      OH[(size_t)(qbase + i) * 512 + h * 64 + d] = (bf16_t)o[vv];
    }
  }
}

// ---------------------------------------------------------------------------
// Elementwise / reduction kernels
// ---------------------------------------------------------------------------
__global__ void __launch_bounds__(256)
k_embed(const float* __restrict__ emb, const int* __restrict__ ids,
        float* __restrict__ X)
{
  int i = blockIdx.x * blockDim.x + threadIdx.x;
  if (i >= 32768 * 512) return;
  int t = i >> 9, d = i & 511;
  X[i] = emb[(size_t)ids[t] * 512 + d];
}

__global__ void __launch_bounds__(256)
k_cvt(const float* __restrict__ src, bf16_t* __restrict__ dst, int n)
{
  int i = blockIdx.x * blockDim.x + threadIdx.x;
  if (i < n) dst[i] = (bf16_t)src[i];
}

// src[K,N] fp32 -> dst[N,Kpad] bf16 (zero pad K..Kpad)
__global__ void __launch_bounds__(256)
k_transpose(const float* __restrict__ src, bf16_t* __restrict__ dst,
            int K, int N, int Kpad)
{
  int i = blockIdx.x * blockDim.x + threadIdx.x;
  if (i >= N * Kpad) return;
  int n = i / Kpad, k = i % Kpad;
  float v = (k < K) ? src[(size_t)k * N + n] : 0.0f;
  dst[i] = (bf16_t)v;
}

// qk-rmsnorm + xpos rotary; QKVC chunk of 4096 tokens -> bf16 Q/K/V (full arrays)
__global__ void __launch_bounds__(256)
k_rotary(const float* __restrict__ qkv, bf16_t* __restrict__ QR,
         bf16_t* __restrict__ KR, bf16_t* __restrict__ VR,
         const float* __restrict__ qs, const float* __restrict__ ks,
         int chunkStart)
{
  int idx = blockIdx.x * blockDim.x + threadIdx.x;
  if (idx >= 4096 * 8) return;
  int tl = idx >> 3, h = idx & 7;
  int tok = chunkStart + tl;
  int p = tok & 4095;  // position within sequence
  const float* base = qkv + (size_t)tl * 1536 + h * 64;

  float q[64], kk[64], qn = 0.0f, kn = 0.0f;
  for (int d = 0; d < 64; d++) {
    q[d]  = base[d];
    kk[d] = base[512 + d];
    qn += q[d] * q[d];
    kn += kk[d] * kk[d];
  }
  qn = fmaxf(sqrtf(qn), 1e-12f);
  kn = fmaxf(sqrtf(kn), 1e-12f);
  float pw = (float)(p - 2048) / 16.0f;  // (t - n/2) / SCALE_BASE

  size_t ob = (size_t)tok * 512 + h * 64;
  for (int d = 0; d < 64; d++) {
    int j = d & 31;
    float inv = powf(10000.0f, -(float)j / 32.0f);
    float fr = (float)p * inv;
    float c = cosf(fr), s = sinf(fr);
    float sv = (2.0f * j + 0.4f * 64.0f) / (1.4f * 64.0f);
    float sc = expf(pw * logf(sv));
    float qv = q[d] / qn * qs[d];
    float kv = kk[d] / kn * ks[d];
    float qr = (d < 32) ? -(q[d + 32] / qn * qs[d + 32]) : (q[d - 32] / qn * qs[d - 32]);
    float kr = (d < 32) ? -(kk[d + 32] / kn * ks[d + 32]) : (kk[d - 32] / kn * ks[d - 32]);
    QR[ob + d] = (bf16_t)((qv * c + qr * s) * sc);
    KR[ob + d] = (bf16_t)((kv * c + kr * s) / sc);
    VR[ob + d] = (bf16_t)base[1024 + d];
  }
}

// GEGLU: HC[row,2730] -> GH[row,1376] bf16  (a * gelu_exact(g), zero-padded K)
__global__ void __launch_bounds__(256)
k_geglu(const float* __restrict__ HC, bf16_t* __restrict__ GH)
{
  int i = blockIdx.x * blockDim.x + threadIdx.x;
  if (i >= 4096 * 1376) return;
  int row = i / 1376, col = i % 1376;
  float val = 0.0f;
  if (col < 1365) {
    float a = HC[(size_t)row * 2730 + col];
    float g = HC[(size_t)row * 2730 + 1365 + col];
    val = a * (0.5f * g * (1.0f + erff(g * 0.70710678f)));
  }
  GH[(size_t)row * 1376 + col] = (bf16_t)val;
}

// per-code squared norms of the codebook
__global__ void __launch_bounds__(256)
k_rowsq(const float* __restrict__ cb, float* __restrict__ cbsq)
{
  int gid = blockIdx.x * blockDim.x + threadIdx.x;
  int lane = gid & 31, row = gid >> 5;
  if (row >= 1024) return;
  float s = 0.0f;
  for (int i = 0; i < 16; i++) {
    float v = cb[(size_t)row * 512 + i * 32 + lane];
    s += v * v;
  }
  s = wave_sum(s);
  if (lane == 0) cbsq[row] = s;
}

// VQ: argmin over codes, gather qz (bf16), accumulate commit loss
__global__ void __launch_bounds__(256)
k_vq(const float* __restrict__ dot, const float* __restrict__ z,
     const float* __restrict__ cb, const float* __restrict__ cbsq,
     bf16_t* __restrict__ qzb, float* __restrict__ commit)
{
  int gid = blockIdx.x * blockDim.x + threadIdx.x;
  int lane = gid & 31, row = gid >> 5;
  if (row >= 8192) return;
  const float* dr = dot + (size_t)row * 1024;
  float best = 3.4e38f; int bi = 0;
  for (int t = 0; t < 32; t++) {
    int c = t * 32 + lane;
    float dv = cbsq[c] - 2.0f * dr[c];
    if (dv < best) { best = dv; bi = c; }
  }
  for (int off = 16; off; off >>= 1) {
    float ob = __shfl_xor(best, off, 32);
    int   oi = __shfl_xor(bi,   off, 32);
    if (ob < best || (ob == best && oi < bi)) { best = ob; bi = oi; }
  }
  const float* crow = cb + (size_t)bi * 512;
  const float* zrow = z  + (size_t)row * 512;
  float sq = 0.0f;
  for (int i = 0; i < 16; i++) {
    int d = i * 32 + lane;
    float qv = crow[d];
    float dz = qv - zrow[d];
    sq += dz * dz;
    qzb[(size_t)row * 512 + d] = (bf16_t)qv;
  }
  sq = wave_sum(sq);
  if (lane == 0) atomicAdd(commit, sq);
}

// final layernorm -> bf16
__global__ void __launch_bounds__(256)
k_ln(const float* __restrict__ X, const float* __restrict__ g,
     const float* __restrict__ bb, bf16_t* __restrict__ XH)
{
  int gid = blockIdx.x * blockDim.x + threadIdx.x;
  int lane = gid & 31, row = gid >> 5;
  if (row >= 32768) return;
  const float* xr = X + (size_t)row * 512;
  float s = 0.0f, ss = 0.0f;
  for (int i = 0; i < 16; i++) {
    float v = xr[i * 32 + lane];
    s += v; ss += v * v;
  }
  s = wave_sum(s); ss = wave_sum(ss);
  float mu = s / 512.0f;
  float var = ss / 512.0f - mu * mu;
  float rstd = rsqrtf(var + 1e-5f);
  for (int i = 0; i < 16; i++) {
    int d = i * 32 + lane;
    XH[(size_t)row * 512 + d] = (bf16_t)((xr[d] - mu) * rstd * g[d] + bb[d]);
  }
}

// cross-entropy accumulation over a 4096-token chunk of logits
__global__ void __launch_bounds__(256)
k_ce(const float* __restrict__ logits, const int* __restrict__ ids,
     float* __restrict__ ce_sum, int chunkStart)
{
  int gid = blockIdx.x * blockDim.x + threadIdx.x;
  int lane = gid & 31, row = gid >> 5;
  if (row >= 4096) return;
  const float* L = logits + (size_t)row * 256;
  float mx = -3.4e38f;
  for (int i = 0; i < 8; i++) mx = fmaxf(mx, L[i * 32 + lane]);
  mx = wave_max(mx);
  float s = 0.0f;
  for (int i = 0; i < 8; i++) s += __expf(L[i * 32 + lane] - mx);
  s = wave_sum(s);
  if (lane == 0) {
    float lse = mx + __logf(s);
    int lbl = ids[chunkStart + row];
    atomicAdd(ce_sum, lse - L[lbl]);
  }
}

__global__ void k_zero(float* s) { s[0] = 0.0f; s[1] = 0.0f; }

__global__ void k_final(const float* __restrict__ s, float* __restrict__ out)
{
  out[0] = s[0] / 32768.0f + s[1] / 67108864.0f;  // ce mean + commit mean
}

// ---------------------------------------------------------------------------
// Host orchestration
// ---------------------------------------------------------------------------
extern "C" void kernel_launch(void* const* d_in, const int* in_sizes, int n_in,
                              void* d_out, int out_size, void* d_ws, size_t ws_size,
                              hipStream_t stream)
{
  (void)in_sizes; (void)n_in; (void)out_size; (void)ws_size;

  const int Mtot = 32768;

  // -------- parameter pointers (setup_inputs dict order) --------
  const float* token_emb = (const float*)d_in[0];
  struct Layer { const float *wqkv, *qs, *ks, *wo, *w1, *b1, *w2, *b2; };
  Layer L[4];
  for (int l = 0; l < 4; l++) {
    int b = 1 + 8 * l;
    L[l].wqkv = (const float*)d_in[b + 0];
    L[l].qs   = (const float*)d_in[b + 1];
    L[l].ks   = (const float*)d_in[b + 2];
    L[l].wo   = (const float*)d_in[b + 3];
    L[l].w1   = (const float*)d_in[b + 4];
    L[l].b1   = (const float*)d_in[b + 5];
    L[l].w2   = (const float*)d_in[b + 6];
    L[l].b2   = (const float*)d_in[b + 7];
  }
  const float* win  = (const float*)d_in[33];
  const float* bin  = (const float*)d_in[34];
  const float* wout = (const float*)d_in[35];
  const float* bout = (const float*)d_in[36];
  const float* cb   = (const float*)d_in[37];
  const float* lng  = (const float*)d_in[38];
  const float* lnb  = (const float*)d_in[39];
  const float* wlog = (const float*)d_in[40];
  const float* blog = (const float*)d_in[41];
  const int*   ids  = (const int*)d_in[42];

  // -------- workspace layout --------
  char* wsb = (char*)d_ws;
  size_t off = 0;
  auto ALLOC = [&](size_t bytes) -> void* {
    off = (off + 255) & ~(size_t)255;
    void* p = wsb + off;
    off += bytes;
    return p;
  };
  bf16_t* wqkvT = (bf16_t*)ALLOC((size_t)4 * 1536 * 512 * 2);
  bf16_t* woT   = (bf16_t*)ALLOC((size_t)4 * 512 * 512 * 2);
  bf16_t* w1T   = (bf16_t*)ALLOC((size_t)4 * 2730 * 512 * 2);
  bf16_t* w2T   = (bf16_t*)ALLOC((size_t)4 * 512 * 1376 * 2);
  bf16_t* winT  = (bf16_t*)ALLOC((size_t)2048 * 512 * 2);
  bf16_t* woutT = (bf16_t*)ALLOC((size_t)512 * 2048 * 2);
  bf16_t* wlogT = (bf16_t*)ALLOC((size_t)256 * 512 * 2);
  bf16_t* cbh   = (bf16_t*)ALLOC((size_t)1024 * 512 * 2);
  float*  cbsq  = (float*) ALLOC((size_t)1024 * 4);
  float*  X     = (float*) ALLOC((size_t)Mtot * 512 * 4);
  bf16_t* XH    = (bf16_t*)ALLOC((size_t)Mtot * 512 * 2);
  float*  QKVC  = (float*) ALLOC((size_t)4096 * 1536 * 4);
  bf16_t* QR    = (bf16_t*)ALLOC((size_t)Mtot * 512 * 2);
  bf16_t* KR    = (bf16_t*)ALLOC((size_t)Mtot * 512 * 2);
  bf16_t* VR    = (bf16_t*)ALLOC((size_t)Mtot * 512 * 2);
  bf16_t* OH    = (bf16_t*)ALLOC((size_t)Mtot * 512 * 2);
  float*  HC    = (float*) ALLOC((size_t)4096 * 2730 * 4);
  bf16_t* GH    = (bf16_t*)ALLOC((size_t)4096 * 1376 * 2);
  float*  ZC    = (float*) ALLOC((size_t)2048 * 2048 * 4);
  bf16_t* ZB    = (bf16_t*)ALLOC((size_t)8192 * 512 * 2);
  float*  DOT   = (float*) ALLOC((size_t)8192 * 1024 * 4);
  bf16_t* QZB   = (bf16_t*)ALLOC((size_t)2048 * 2048 * 2);
  float*  LOGB  = (float*) ALLOC((size_t)4096 * 256 * 4);
  float*  scal  = (float*) ALLOC(256);  // [0]=ce_sum, [1]=commit_sum

  auto gemm = [&](const bf16_t* A, const bf16_t* Bt, float* C, const float* bias,
                  int M, int N, int K, int lda, int ldb, int ldc, int res) {
    dim3 g((N + 127) / 128, M / 64);
    k_gemm<<<g, dim3(256), 0, stream>>>(A, Bt, C, bias, N, K, lda, ldb, ldc, res);
  };
  auto cvt = [&](const float* s, bf16_t* d, int n) {
    k_cvt<<<(n + 255) / 256, 256, 0, stream>>>(s, d, n);
  };
  auto tr = [&](const float* s, bf16_t* d, int K, int N, int Kpad) {
    k_transpose<<<(N * Kpad + 255) / 256, 256, 0, stream>>>(s, d, K, N, Kpad);
  };

  // -------- prep: weight transposes, codebook, accumulators, embedding -----
  k_zero<<<1, 1, 0, stream>>>(scal);
  for (int l = 0; l < 4; l++) {
    tr(L[l].wqkv, wqkvT + (size_t)l * 1536 * 512, 512, 1536, 512);
    tr(L[l].wo,   woT   + (size_t)l * 512 * 512,  512, 512,  512);
    tr(L[l].w1,   w1T   + (size_t)l * 2730 * 512, 512, 2730, 512);
    tr(L[l].w2,   w2T   + (size_t)l * 512 * 1376, 1365, 512, 1376);
  }
  tr(win,  winT,  512, 2048, 512);
  tr(wout, woutT, 2048, 512, 2048);
  tr(wlog, wlogT, 512, 256, 512);
  cvt(cb, cbh, 1024 * 512);
  k_rowsq<<<(1024 * 32) / 256, 256, 0, stream>>>(cb, cbsq);
  k_embed<<<(Mtot * 512) / 256, 256, 0, stream>>>(token_emb, ids, X);

  // -------- one transformer layer (attention + GEGLU FF, both residual) ----
  auto layer = [&](int l) {
    cvt(X, XH, Mtot * 512);
    for (int c = 0; c < 8; c++) {
      gemm(XH + (size_t)c * 4096 * 512, wqkvT + (size_t)l * 1536 * 512,
           QKVC, nullptr, 4096, 1536, 512, 512, 512, 1536, 0);
      k_rotary<<<(4096 * 8) / 256, 256, 0, stream>>>(
          QKVC, QR, KR, VR, L[l].qs, L[l].ks, c * 4096);
    }
    k_attention<<<dim3(128, 8, 8), 256, 0, stream>>>(QR, KR, VR, OH);
    gemm(OH, woT + (size_t)l * 512 * 512, X, nullptr,
         Mtot, 512, 512, 512, 512, 512, /*residual*/1);
    cvt(X, XH, Mtot * 512);
    for (int c = 0; c < 8; c++) {
      gemm(XH + (size_t)c * 4096 * 512, w1T + (size_t)l * 2730 * 512,
           HC, L[l].b1, 4096, 2730, 512, 512, 512, 2730, 0);
      k_geglu<<<(4096 * 1376 + 255) / 256, 256, 0, stream>>>(HC, GH);
      gemm(GH, w2T + (size_t)l * 512 * 1376, X + (size_t)c * 4096 * 512,
           L[l].b2, 4096, 512, 1376, 1376, 1376, 512, /*residual*/1);
    }
  };

  // -------- encoder --------
  for (int l = 0; l < 4; l++) layer(l);

  // -------- vector quantizer --------
  cvt(X, XH, Mtot * 512);
  for (int c = 0; c < 16; c++) {
    gemm(XH + (size_t)c * 2048 * 512, winT, ZC, bin,
         2048, 2048, 512, 512, 512, 2048, 0);
    cvt(ZC, ZB, 2048 * 2048);
    gemm(ZB, cbh, DOT, nullptr, 8192, 1024, 512, 512, 512, 1024, 0);
    k_vq<<<(8192 * 32) / 256, 256, 0, stream>>>(DOT, ZC, cb, cbsq, QZB, scal + 1);
    gemm(QZB, woutT, X + (size_t)c * 2048 * 512, bout,
         2048, 512, 2048, 2048, 2048, 512, 0);
  }

  // -------- decoder (reuses encoder layers) --------
  for (int l = 0; l < 4; l++) layer(l);

  // -------- final LN + logits + CE --------
  k_ln<<<(Mtot * 32) / 256, 256, 0, stream>>>(X, lng, lnb, XH);
  for (int c = 0; c < 8; c++) {
    gemm(XH + (size_t)c * 4096 * 512, wlogT, LOGB, blog,
         4096, 256, 512, 512, 512, 256, 0);
    k_ce<<<(4096 * 32) / 256, 256, 0, stream>>>(LOGB, ids, scal, c * 4096);
  }
  k_final<<<1, 1, 0, stream>>>(scal, (float*)d_out);
}
